// VQCodebook_13142599926205
// MI455X (gfx1250) — matrix operand order repaired and checked by reading, not compiled, not run on hardware
//
#include <hip/hip_runtime.h>
#include <math.h>

#define WG 256
#define LSTR 40  // LDS row stride in bf16 elements: 32 data + 8 pad (80B rows, 16B aligned)

typedef __attribute__((ext_vector_type(4)))  unsigned short v4u;
typedef __attribute__((ext_vector_type(8)))  unsigned short v8u;
typedef __attribute__((ext_vector_type(16))) unsigned short v16u;
typedef __attribute__((ext_vector_type(16))) __bf16        v16bf;
typedef __attribute__((ext_vector_type(8)))  float         v8f;

// ---------- helpers ----------

__device__ __forceinline__ unsigned short f2bf(float x) {
  unsigned u = __float_as_uint(x);
  return (unsigned short)((u + 0x7FFFu + ((u >> 16) & 1u)) >> 16);  // RNE
}

// Stage a 128-row x 32-col fp32 tile into LDS as bf16. ld = row stride (elems).
// If pf, first issue one prefetch per thread for the tile two k-steps ahead
// (hoisted out of the copy loop: no EXEC-mask churn in the inner loop).
__device__ __forceinline__ void stage_tile(const float* __restrict__ G, int ld, int row0,
                                           int k0, unsigned short* __restrict__ S, bool pf) {
  if (pf) {
    int r = threadIdx.x >> 1;                 // 0..127: one prefetch per 64B half-row
    int h = (threadIdx.x & 1) * 16;
    __builtin_prefetch(G + (size_t)(row0 + r) * ld + (k0 + 64 + h), 0, 2);
  }
  for (int c = threadIdx.x; c < 128 * 8; c += WG) {
    int r = c >> 3, q = c & 7;
    float4 f = *(const float4*)(G + (size_t)(row0 + r) * ld + (k0 + q * 4));
    v4u p;
    p[0] = f2bf(f.x); p[1] = f2bf(f.y); p[2] = f2bf(f.z); p[3] = f2bf(f.w);
    *(v4u*)(S + r * LSTR + q * 4) = p;
  }
}

// A fragment (16x32 MxK): lane = (m, h); elems 0..7 -> K=8h+e, elems 8..15 -> K=16+8h+(e-8)
__device__ __forceinline__ v16u load_fragA(const unsigned short* S, int rowbase) {
  int l = threadIdx.x & 31, lm = l & 15, lh = l >> 4;
  const unsigned short* p = S + (rowbase + lm) * LSTR + 8 * lh;
  v8u lo = *(const v8u*)p;
  v8u hi = *(const v8u*)(p + 16);
  v16u r;
#pragma unroll
  for (int i = 0; i < 8; ++i) { r[i] = lo[i]; r[i + 8] = hi[i]; }
  return r;
}

// B fragment (32x16 KxN) from row-major [N,K]: lane = (n, h); elems 0..7 -> K=8h+e, elems 8..15 -> K=16+8h+(e-8)
__device__ __forceinline__ v16u load_fragB(const unsigned short* S, int rowbase) {
  int l = threadIdx.x & 31, lm = l & 15, lh = l >> 4;
  const unsigned short* p = S + (rowbase + lm) * LSTR + 8 * lh;
  v8u lo = *(const v8u*)p;
  v8u hi = *(const v8u*)(p + 16);
  v16u r;
#pragma unroll
  for (int i = 0; i < 8; ++i) { r[i] = lo[i]; r[i + 8] = hi[i]; }
  return r;
}

// C = A[M,K] * B[N,K]^T for one 128x128 block, double-buffered LDS pipeline.
// WMMAs are issued BEFORE staging the next tile so the 6 fragment registers are
// dead during staging (keeps VGPR pressure below spill threshold); the matrix
// pipe co-executes with the staging VMEM/DS traffic.
// acc[i][j] elem r of lane l:
//   row_local = wm*32 + i*16 + 8*(l>>4) + r ; col_local = wn*64 + j*16 + (l&15)
__device__ __forceinline__ void gemm128(const float* __restrict__ A, int lda,
                                        const float* __restrict__ B, int ldb, int K,
                                        int row0, int col0,
                                        unsigned short* As0, unsigned short* Bs0,
                                        unsigned short* As1, unsigned short* Bs1,
                                        v8f acc[2][4]) {
  int wave = threadIdx.x >> 5, wm = wave & 3, wn = wave >> 2;
#pragma unroll
  for (int i = 0; i < 2; ++i)
#pragma unroll
    for (int j = 0; j < 4; ++j) acc[i][j] = (v8f){0, 0, 0, 0, 0, 0, 0, 0};

  stage_tile(A, lda, row0, 0, As0, true);
  stage_tile(B, ldb, col0, 0, Bs0, true);
  __syncthreads();

  int nk = K / 32;
  for (int t = 0; t < nk; ++t) {
    const unsigned short* Ac = (t & 1) ? As1 : As0;
    const unsigned short* Bc = (t & 1) ? Bs1 : Bs0;
    // fragments for current tile
    v16u af[2], bfr[4];
    af[0] = load_fragA(Ac, wm * 32);
    af[1] = load_fragA(Ac, wm * 32 + 16);
#pragma unroll
    for (int j = 0; j < 4; ++j) bfr[j] = load_fragB(Bc, wn * 64 + j * 16);
    // matrix ops first: fragment regs die here, acc stays resident
#pragma unroll
    for (int i = 0; i < 2; ++i)
#pragma unroll
      for (int j = 0; j < 4; ++j)
        acc[i][j] = __builtin_amdgcn_wmma_f32_16x16x32_bf16(
            false, __builtin_bit_cast(v16bf, af[i]),
            false, __builtin_bit_cast(v16bf, bfr[j]),
            (short)0, acc[i][j], false, false);
    // stage next tile into the alternate buffer while WMMAs drain
    if (t + 1 < nk) {
      stage_tile(A, lda, row0, (t + 1) * 32, (t & 1) ? As0 : As1, t + 2 < nk);
      stage_tile(B, ldb, col0, (t + 1) * 32, (t & 1) ? Bs0 : Bs1, t + 2 < nk);
    }
    __syncthreads();  // single barrier per k-step
  }
}

#define GEMM_SHARED                               \
  __shared__ unsigned short As0[128 * LSTR];      \
  __shared__ unsigned short Bs0[128 * LSTR];      \
  __shared__ unsigned short As1[128 * LSTR];      \
  __shared__ unsigned short Bs1[128 * LSTR];

__device__ __forceinline__ float block_reduce_sum(float v) {
  __shared__ float red[8];
#pragma unroll
  for (int d = 16; d >= 1; d >>= 1) v += __shfl_xor(v, d);
  int w = threadIdx.x >> 5;
  __syncthreads();
  if ((threadIdx.x & 31) == 0) red[w] = v;
  __syncthreads();
  float t = 0.f;
  if (threadIdx.x == 0) {
    int nw = (blockDim.x + 31) >> 5;
    for (int i = 0; i < nw; ++i) t += red[i];
  }
  return t;  // valid on thread 0
}

// ---------- kernels ----------

__global__ void vq_init_kernel(float* scal, float* avgA, float* avgB) {
  int i = blockIdx.x * WG + threadIdx.x;
  if (i < 8) scal[i] = 0.f;
  if (i < 4096) { avgA[i] = 0.f; avgB[i] = 0.f; }
}

// Y = X @ W^T + bias ; X:[8192,Kdim], W:[512,Kdim], Y:[8192,512]
__global__ void __launch_bounds__(WG, 1)
vq_proj_kernel(const float* __restrict__ X, const float* __restrict__ W,
               const float* __restrict__ bias, float* __restrict__ Y, int Kdim) {
  GEMM_SHARED
  int row0 = blockIdx.y * 128, col0 = blockIdx.x * 128;
  v8f acc[2][4];
  gemm128(X, Kdim, W, Kdim, Kdim, row0, col0, As0, Bs0, As1, Bs1, acc);
  int wave = threadIdx.x >> 5, wm = wave & 3, wn = wave >> 2;
  int l = threadIdx.x & 31, lm = l & 15, lh = l >> 4;
#pragma unroll
  for (int i = 0; i < 2; ++i)
#pragma unroll
    for (int j = 0; j < 4; ++j) {
      int col = col0 + wn * 64 + j * 16 + lm;
      float bv = bias[col];
#pragma unroll
      for (int r = 0; r < 8; ++r) {
        int row = row0 + wm * 32 + i * 16 + lh * 8 + r;
        Y[(size_t)row * 512 + col] = acc[i][j][r] + bv;
      }
    }
}

// Z = Y / max(||Y||_row, 1e-12), one block per row, D=512
__global__ void vq_normalize_kernel(const float* __restrict__ Y, float* __restrict__ Z) {
  int row = blockIdx.x;
  const float* y = Y + (size_t)row * 512;
  float ss = 0.f;
  for (int c = threadIdx.x; c < 512; c += WG) { float v = y[c]; ss += v * v; }
  float tot = block_reduce_sum(ss);
  __shared__ float sinv;
  if (threadIdx.x == 0) sinv = 1.f / fmaxf(sqrtf(tot), 1e-12f);
  __syncthreads();
  float inv = sinv;
  for (int c = threadIdx.x; c < 512; c += WG) Z[(size_t)row * 512 + c] = y[c] * inv;
}

// c2[k] = ||codebook[k]||^2 ; one wave per row
__global__ void vq_rownorm2_kernel(const float* __restrict__ C, float* __restrict__ c2) {
  int row = blockIdx.x * 8 + (threadIdx.x >> 5);
  const float* c = C + (size_t)row * 512;
  float ss = 0.f;
  for (int i = threadIdx.x & 31; i < 512; i += 32) { float v = c[i]; ss += v * v; }
#pragma unroll
  for (int d = 16; d >= 1; d >>= 1) ss += __shfl_xor(ss, d);
  if ((threadIdx.x & 31) == 0) c2[row] = ss;
}

// Pass 1: S = Z @ CB^T ; val = -5*sqrt(max(1 + c2 - 2S, 0)). Write per-(row, 64-col block)
// partials (max, sum-exp-rel, argmax col). Partial grid: 64 col-blocks.
__global__ void __launch_bounds__(WG, 1)
vq_cdist_pass1(const float* __restrict__ Z, const float* __restrict__ CB,
               const float* __restrict__ c2, float* __restrict__ pm,
               float* __restrict__ ps, int* __restrict__ pi) {
  GEMM_SHARED
  int row0 = blockIdx.y * 128, col0 = blockIdx.x * 128;
  v8f acc[2][4];
  gemm128(Z, 512, CB, 512, 512, row0, col0, As0, Bs0, As1, Bs1, acc);
  int wave = threadIdx.x >> 5, wm = wave & 3, wn = wave >> 2;
  int l = threadIdx.x & 31, lm = l & 15, lh = l >> 4;
  int cbk = blockIdx.x * 2 + wn;
  int colbase = col0 + wn * 64;
  float c2v[4];
#pragma unroll
  for (int j = 0; j < 4; ++j) c2v[j] = c2[colbase + j * 16 + lm];
#pragma unroll
  for (int i = 0; i < 2; ++i)
#pragma unroll
    for (int r = 0; r < 8; ++r) {
      float vals[4];
      float m = -INFINITY;
      int idx = 0x7FFFFFFF;
#pragma unroll
      for (int j = 0; j < 4; ++j) {
        float s = acc[i][j][r];
        float d2 = fmaxf(1.0f + c2v[j] - 2.0f * s, 0.0f);
        float v = -5.0f * sqrtf(d2);
        vals[j] = v;
        int cc = colbase + j * 16 + lm;
        if (v > m || (v == m && cc < idx)) { m = v; idx = cc; }
      }
#pragma unroll
      for (int d = 1; d < 16; d <<= 1) {
        float om = __shfl_xor(m, d);
        int oi = __shfl_xor(idx, d);
        if (om > m || (om == m && oi < idx)) { m = om; idx = oi; }
      }
      float s = 0.f;
#pragma unroll
      for (int j = 0; j < 4; ++j) s += __expf(vals[j] - m);
#pragma unroll
      for (int d = 1; d < 16; d <<= 1) s += __shfl_xor(s, d);
      if (lm == 0) {
        int row = row0 + wm * 32 + i * 16 + lh * 8 + r;
        size_t off = (size_t)row * 64 + cbk;
        pm[off] = m; ps[off] = s; pi[off] = idx;
      }
    }
}

// Combine 64 (m,s,idx) partials per row -> argmin index + logsumexp (in val space)
__global__ void vq_cdist_combine(const float* __restrict__ pm, const float* __restrict__ ps,
                                 const int* __restrict__ pi, int* __restrict__ idx_out,
                                 float* __restrict__ logZ_out) {
  int i = blockIdx.x * WG + threadIdx.x;  // exactly 8192
  float gm = -INFINITY, gs = 0.f;
  int gi = 0;
  for (int p = 0; p < 64; ++p) {
    size_t off = (size_t)i * 64 + p;
    float m = pm[off], s = ps[off];
    int id = pi[off];
    if (m > gm) { gs = gs * __expf(gm - m) + s; gm = m; gi = id; }
    else        { gs += s * __expf(m - gm); }
  }
  idx_out[i] = gi;
  logZ_out[i] = gm + __logf(gs);
}

// Pass 2: recompute, accumulate column sums of normalized softmax probs into avgacc
__global__ void __launch_bounds__(WG, 1)
vq_cdist_pass2(const float* __restrict__ Z, const float* __restrict__ CB,
               const float* __restrict__ c2, const float* __restrict__ logZ,
               float* __restrict__ avgacc) {
  GEMM_SHARED
  int row0 = blockIdx.y * 128, col0 = blockIdx.x * 128;
  v8f acc[2][4];
  gemm128(Z, 512, CB, 512, 512, row0, col0, As0, Bs0, As1, Bs1, acc);
  int wave = threadIdx.x >> 5, wm = wave & 3, wn = wave >> 2;
  int l = threadIdx.x & 31, lm = l & 15, lh = l >> 4;
  int colbase = col0 + wn * 64;
  float c2v[4];
#pragma unroll
  for (int j = 0; j < 4; ++j) c2v[j] = c2[colbase + j * 16 + lm];
  float colacc[4] = {0.f, 0.f, 0.f, 0.f};
#pragma unroll
  for (int i = 0; i < 2; ++i)
#pragma unroll
    for (int r = 0; r < 8; ++r) {
      int row = row0 + wm * 32 + i * 16 + lh * 8 + r;
      float lz = logZ[row];
#pragma unroll
      for (int j = 0; j < 4; ++j) {
        float s = acc[i][j][r];
        float d2 = fmaxf(1.0f + c2v[j] - 2.0f * s, 0.0f);
        float v = -5.0f * sqrtf(d2);
        colacc[j] += __expf(v - lz);
      }
    }
#pragma unroll
  for (int j = 0; j < 4; ++j) colacc[j] += __shfl_xor(colacc[j], 16);
  if (lh == 0)
#pragma unroll
    for (int j = 0; j < 4; ++j) atomicAdd(&avgacc[colbase + j * 16 + lm], colacc[j]);
}

// sim = ZA @ ZB^T / 0.07 ; per-row partials over 128 col-blocks-of-64,
// per-col partials over 256 row-blocks-of-32
__global__ void __launch_bounds__(WG, 1)
vq_sim_kernel(const float* __restrict__ ZA, const float* __restrict__ ZB,
              float* __restrict__ pmr, float* __restrict__ psr,
              float* __restrict__ pmc, float* __restrict__ psc) {
  GEMM_SHARED
  int row0 = blockIdx.y * 128, col0 = blockIdx.x * 128;
  v8f acc[2][4];
  gemm128(ZA, 512, ZB, 512, 512, row0, col0, As0, Bs0, As1, Bs1, acc);
  const float invT = 1.0f / 0.07f;
  int wave = threadIdx.x >> 5, wm = wave & 3, wn = wave >> 2;
  int l = threadIdx.x & 31, lm = l & 15, lh = l >> 4;
  // row direction
#pragma unroll
  for (int i = 0; i < 2; ++i)
#pragma unroll
    for (int r = 0; r < 8; ++r) {
      float vals[4], m = -INFINITY;
#pragma unroll
      for (int j = 0; j < 4; ++j) { vals[j] = acc[i][j][r] * invT; m = fmaxf(m, vals[j]); }
#pragma unroll
      for (int d = 1; d < 16; d <<= 1) m = fmaxf(m, __shfl_xor(m, d));
      float s = 0.f;
#pragma unroll
      for (int j = 0; j < 4; ++j) s += __expf(vals[j] - m);
#pragma unroll
      for (int d = 1; d < 16; d <<= 1) s += __shfl_xor(s, d);
      if (lm == 0) {
        int row = row0 + wm * 32 + i * 16 + lh * 8 + r;
        size_t off = (size_t)row * 128 + blockIdx.x * 2 + wn;
        pmr[off] = m; psr[off] = s;
      }
    }
  // col direction
#pragma unroll
  for (int j = 0; j < 4; ++j) {
    float m = -INFINITY;
#pragma unroll
    for (int i = 0; i < 2; ++i)
#pragma unroll
      for (int r = 0; r < 8; ++r) m = fmaxf(m, acc[i][j][r] * invT);
    m = fmaxf(m, __shfl_xor(m, 16));
    float s = 0.f;
#pragma unroll
    for (int i = 0; i < 2; ++i)
#pragma unroll
      for (int r = 0; r < 8; ++r) s += __expf(acc[i][j][r] * invT - m);
    s += __shfl_xor(s, 16);
    if (lh == 0) {
      int col = col0 + wn * 64 + j * 16 + lm;
      size_t off = (size_t)col * 256 + blockIdx.y * 4 + wm;
      pmc[off] = m; psc[off] = s;
    }
  }
}

__global__ void vq_combine_lse(const float* __restrict__ pm, const float* __restrict__ ps,
                               int nparts, float* __restrict__ logZ) {
  int i = blockIdx.x * WG + threadIdx.x;  // exactly 8192
  float gm = -INFINITY, gs = 0.f;
  for (int p = 0; p < nparts; ++p) {
    size_t off = (size_t)i * nparts + p;
    float m = pm[off], s = ps[off];
    if (m > gm) { gs = gs * __expf(gm - m) + s; gm = m; }
    else        { gs += s * __expf(m - gm); }
  }
  logZ[i] = gm + __logf(gs);
}

// diag[i] = dot(za_i, zb_i)/0.07 ; one wave per row
__global__ void vq_diag_kernel(const float* __restrict__ ZA, const float* __restrict__ ZB,
                               float* __restrict__ dg) {
  int row = blockIdx.x * 8 + (threadIdx.x >> 5);
  const float* a = ZA + (size_t)row * 512;
  const float* b = ZB + (size_t)row * 512;
  float s = 0.f;
  for (int i = threadIdx.x & 31; i < 512; i += 32) s += a[i] * b[i];
#pragma unroll
  for (int d = 16; d >= 1; d >>= 1) s += __shfl_xor(s, d);
  if ((threadIdx.x & 31) == 0) dg[row] = s * (1.0f / 0.07f);
}

// sum of (codebook[idx[row]] - z[row])^2 -> atomicAdd into acc; one block per row
__global__ void vq_rec_kernel(const float* __restrict__ Z, const float* __restrict__ CB,
                              const int* __restrict__ idx, float* __restrict__ acc) {
  int row = blockIdx.x;
  const float* z = Z + (size_t)row * 512;
  const float* q = CB + (size_t)idx[row] * 512;
  float s = 0.f;
  for (int c = threadIdx.x; c < 512; c += WG) { float d = q[c] - z[c]; s += d * d; }
  float t = block_reduce_sum(s);
  if (threadIdx.x == 0) atomicAdd(acc, t);
}

// entropy term: acc += -sum(avg * log(avg + eps)), avg = avgacc/8192
__global__ void vq_div_kernel(const float* __restrict__ avgacc, float* __restrict__ dacc) {
  int i = blockIdx.x * WG + threadIdx.x;  // exactly 4096
  float avg = avgacc[i] * (1.0f / 8192.0f);
  float t = -avg * __logf(avg + 1e-8f);
  float s = block_reduce_sum(t);
  if (threadIdx.x == 0) atomicAdd(dacc, s);
}

__global__ void vq_match_kernel(const int* __restrict__ ia, const int* __restrict__ ib,
                                float* __restrict__ macc) {
  int i = blockIdx.x * WG + threadIdx.x;  // exactly 8192
  float t = (ia[i] == ib[i]) ? 1.f : 0.f;
  float s = block_reduce_sum(t);
  if (threadIdx.x == 0) atomicAdd(macc, s);
}

__global__ void vq_cm_kernel(const float* __restrict__ lzr, const float* __restrict__ lzc,
                             const float* __restrict__ dg, float* __restrict__ ab,
                             float* __restrict__ ba) {
  int i = blockIdx.x * WG + threadIdx.x;  // exactly 8192
  float t1 = lzr[i] - dg[i];
  float t2 = lzc[i] - dg[i];
  float s1 = block_reduce_sum(t1);
  if (threadIdx.x == 0) atomicAdd(ab, s1);
  float s2 = block_reduce_sum(t2);
  if (threadIdx.x == 0) atomicAdd(ba, s2);
}

// scal: [0]=recSum [1]=divAcc [2]=matchCnt [3]=cmAB [4]=cmBA
__global__ void vq_final_kernel(const float* __restrict__ sc, float* __restrict__ out) {
  if (threadIdx.x == 0) {
    float rec = 1.25f * sc[0] / (8192.0f * 512.0f);
    float cm = 0.5f * (sc[3] + sc[4]) / 8192.0f;
    float dv = 0.5f * sc[1];
    out[0] = rec + 0.5f * cm - 0.1f * dv;
    out[1] = sc[2] / 8192.0f;
  }
}

// ---------- host ----------

extern "C" void kernel_launch(void* const* d_in, const int* in_sizes, int n_in,
                              void* d_out, int out_size, void* d_ws, size_t ws_size,
                              hipStream_t stream) {
  (void)in_sizes; (void)n_in; (void)out_size; (void)ws_size;
  const float* a  = (const float*)d_in[0];
  const float* b  = (const float*)d_in[1];
  const float* Wa = (const float*)d_in[2];
  const float* ba = (const float*)d_in[3];
  const float* Wb = (const float*)d_in[4];
  const float* bb = (const float*)d_in[5];
  const float* cb = (const float*)d_in[6];

  const int N = 8192, D = 512, NC = 4096;
  char* w = (char*)d_ws;
  size_t off = 0;
  auto alloc = [&](size_t elems) {
    void* p = w + off;
    off += (elems * 4 + 255) & ~(size_t)255;
    return p;
  };
  float* ya   = (float*)alloc((size_t)N * D);
  float* yb   = (float*)alloc((size_t)N * D);
  float* za   = (float*)alloc((size_t)N * D);
  float* zb   = (float*)alloc((size_t)N * D);
  float* c2   = (float*)alloc(NC);
  float* pm   = (float*)alloc((size_t)N * 64);   // reused for modality a then b
  float* ps   = (float*)alloc((size_t)N * 64);
  int*   pi   = (int*)  alloc((size_t)N * 64);
  int*   idxA = (int*)  alloc(N);
  int*   idxB = (int*)  alloc(N);
  float* lzA  = (float*)alloc(N);
  float* lzB  = (float*)alloc(N);
  float* avgA = (float*)alloc(NC);
  float* avgB = (float*)alloc(NC);
  float* pmr  = (float*)alloc((size_t)N * 128);
  float* psr  = (float*)alloc((size_t)N * 128);
  float* pmc  = (float*)alloc((size_t)N * 256);
  float* psc  = (float*)alloc((size_t)N * 256);
  float* lzR  = (float*)alloc(N);
  float* lzC  = (float*)alloc(N);
  float* dg   = (float*)alloc(N);
  float* scal = (float*)alloc(16);

  dim3 blk(WG);

  vq_init_kernel<<<16, blk, 0, stream>>>(scal, avgA, avgB);

  // projections + normalize
  vq_proj_kernel<<<dim3(4, 64), blk, 0, stream>>>(a, Wa, ba, ya, 1024);
  vq_proj_kernel<<<dim3(4, 64), blk, 0, stream>>>(b, Wb, bb, yb, 1024);
  vq_normalize_kernel<<<N, blk, 0, stream>>>(ya, za);
  vq_normalize_kernel<<<N, blk, 0, stream>>>(yb, zb);
  vq_rownorm2_kernel<<<NC / 8, blk, 0, stream>>>(cb, c2);

  // cdist + soft assignment, modality A (partials reused afterwards for B)
  vq_cdist_pass1<<<dim3(32, 64), blk, 0, stream>>>(za, cb, c2, pm, ps, pi);
  vq_cdist_combine<<<N / WG, blk, 0, stream>>>(pm, ps, pi, idxA, lzA);
  vq_cdist_pass2<<<dim3(32, 64), blk, 0, stream>>>(za, cb, c2, lzA, avgA);
  // modality B
  vq_cdist_pass1<<<dim3(32, 64), blk, 0, stream>>>(zb, cb, c2, pm, ps, pi);
  vq_cdist_combine<<<N / WG, blk, 0, stream>>>(pm, ps, pi, idxB, lzB);
  vq_cdist_pass2<<<dim3(32, 64), blk, 0, stream>>>(zb, cb, c2, lzB, avgB);

  // contrastive similarity: one GEMM pass, both softmax directions
  vq_sim_kernel<<<dim3(64, 64), blk, 0, stream>>>(za, zb, pmr, psr, pmc, psc);
  vq_combine_lse<<<N / WG, blk, 0, stream>>>(pmr, psr, 128, lzR);
  vq_combine_lse<<<N / WG, blk, 0, stream>>>(pmc, psc, 256, lzC);
  vq_diag_kernel<<<N / 8, blk, 0, stream>>>(za, zb, dg);

  // loss pieces
  vq_rec_kernel<<<N, blk, 0, stream>>>(za, cb, idxA, scal + 0);
  vq_rec_kernel<<<N, blk, 0, stream>>>(zb, cb, idxB, scal + 0);
  vq_div_kernel<<<NC / WG, blk, 0, stream>>>(avgA, scal + 1);
  vq_div_kernel<<<NC / WG, blk, 0, stream>>>(avgB, scal + 1);
  vq_match_kernel<<<N / WG, blk, 0, stream>>>(idxA, idxB, scal + 2);
  vq_cm_kernel<<<N / WG, blk, 0, stream>>>(lzR, lzC, dg, scal + 3, scal + 4);

  vq_final_kernel<<<1, 32, 0, stream>>>(scal, (float*)d_out);
}
